// RiskyObject_26293789787005
// MI455X (gfx1250) — matrix operand order repaired and checked by reading, not compiled, not run on hardware
//
#include <hip/hip_runtime.h>
#include <hip/hip_bf16.h>

typedef __attribute__((ext_vector_type(16))) __bf16 v16bf;
typedef __attribute__((ext_vector_type(8)))  __bf16 v8bf;
typedef __attribute__((ext_vector_type(8)))  float  v8f;
typedef __attribute__((ext_vector_type(4)))  unsigned int u32x4;
typedef __attribute__((ext_vector_type(8)))  int i32x8;
typedef __attribute__((ext_vector_type(4)))  int i32x4;

#define T_STEPS 191
#define NBX   30
#define HID   256
#define COR   32
#define FDIM  4096
#define NPOV  31   // NB+1
#define KCH   256  // K-chunk staged per TDM transfer in rgb_gemm

// ---- workspace layout (float units) ----
#define OFF_RGB 0
#define SZ_RGB  (T_STEPS*NPOV*HID)          // 5921*256
#define OFF_GX  (OFF_RGB + SZ_RGB)
#define SZ_GX   (T_STEPS*NBX*3*HID)         // 5730*768
#define OFF_BF  (OFF_GX + SZ_GX)
#define SZ_BF   (T_STEPS*NBX*HID)
#define OFF_GXC (OFF_BF + SZ_BF)
#define SZ_GXC  (T_STEPS*NBX*3*COR)
#define OFF_WT  (((OFF_GXC + SZ_GXC) + 15) & ~15)   // 64B aligned bf16 region
// bf16 element offsets inside WT region
#define WB_RGBT 0                            // 256 x 4096
#define WB_XGT  (FDIM*HID)                   // 768 x 512
#define WB_HGT  (WB_XGT + (3*HID)*(2*HID))   // 768 x 256

static __device__ __forceinline__ float sigf(float x) {
  return 1.f / (1.f + __expf(-x));
}

static __device__ __forceinline__ v16bf cat8(v8bf lo, v8bf hi) {
  return __builtin_shufflevector(lo, hi, 0, 1, 2, 3, 4, 5, 6, 7,
                                         8, 9, 10, 11, 12, 13, 14, 15);
}

// ---------------------------------------------------------------------------
// TDM: async 2D tile (rows x KCH f32) global -> LDS with bank padding.
// D# per CDNA5 ISA: group0 = {flags, lds_addr, global_addr, type},
// group1 = dims/strides. pad: interval=256 dwords (code 7), amount=2 dwords
// (code 1) -> LDS row stride = 258 dwords.
// ---------------------------------------------------------------------------
static __device__ __forceinline__
void tdm_load_tile(unsigned long long gaddr, unsigned int ldsaddr,
                   unsigned int rows_valid) {
  u32x4 g0; i32x8 g1;
  i32x4 gz4 = {0, 0, 0, 0};
  i32x8 gz8 = {0, 0, 0, 0, 0, 0, 0, 0};
  g0[0] = 1u;                                   // count=1
  g0[1] = ldsaddr;                              // LDS byte address
  g0[2] = (unsigned int)gaddr;                  // global_addr[31:0]
  g0[3] = (unsigned int)((gaddr >> 32) & 0x1FFFFFFu) | (2u << 30); // type=2
  const unsigned int td0 = FDIM;                // tensor_dim0 (elements)
  const unsigned int td1 = rows_valid;          // rows beyond -> zero fill
  g1[0] = (int)0x3D20000u;   // data_size=2(4B) | pad_en | interval=7 | amount=1
  g1[1] = (int)((td0 & 0xFFFFu) << 16);
  g1[2] = (int)(((td0 >> 16) & 0xFFFFu) | ((td1 & 0xFFFFu) << 16));
  g1[3] = (int)(((td1 >> 16) & 0xFFFFu) | ((unsigned int)KCH << 16)); // tile_dim0
  g1[4] = 16;                                   // tile_dim1=16, tile_dim2=0
  g1[5] = (int)FDIM;                            // tensor_dim0_stride (elements)
  g1[6] = 0;
  g1[7] = 0;
  __builtin_amdgcn_tensor_load_to_lds(g0, g1, gz4, gz4, gz8, 0);
}

// ---------------------------------------------------------------------------
// Kernel 0: transpose+convert weights to bf16 (N-major) once.
// ---------------------------------------------------------------------------
__global__ __launch_bounds__(256)
void prep_weights(const float* __restrict__ Wrgb, const float* __restrict__ Wxg,
                  const float* __restrict__ Whg, __bf16* __restrict__ WT) {
  const int idx = blockIdx.x * 256 + threadIdx.x;
  const int n1 = HID * FDIM;                 // 1048576
  const int n2 = n1 + (3 * HID) * (2 * HID); // + 393216
  const int n3 = n2 + (3 * HID) * HID;       // + 196608
  if (idx < n1) {
    const int n = idx / FDIM, k = idx % FDIM;
    WT[WB_RGBT + idx] = (__bf16)Wrgb[(size_t)k * HID + n];
  } else if (idx < n2) {
    const int r = idx - n1;
    const int n = r / (2 * HID), k = r % (2 * HID);
    WT[WB_XGT + r] = (__bf16)Wxg[(size_t)k * (3 * HID) + n];
  } else if (idx < n3) {
    const int r = idx - n2;
    const int n = r / HID, k = r % HID;
    WT[WB_HGT + r] = (__bf16)Whg[(size_t)k * (3 * HID) + n];
  }
}

// ---------------------------------------------------------------------------
// Kernel 1: RGB = relu(flow[0:T] @ W_rgb + b_rgb)   (M=5921, K=4096, N=256)
// One block per 16-row M-tile; A panel staged by TDM (double buffered);
// 8 waves x 2 N-tiles; B = pre-transposed bf16, contiguous 16B loads.
// ---------------------------------------------------------------------------
__global__ __launch_bounds__(256)
void rgb_gemm(const float* __restrict__ flow, const __bf16* __restrict__ WrgbT,
              const float* __restrict__ brgb, float* __restrict__ RGB) {
  __shared__ float As[2][16 * 258 + 2];   // padded TDM destination, 2 buffers
  const int M = T_STEPS * NPOV;
  const int tid = threadIdx.x;
  const int wave = tid >> 5, lane = tid & 31;
  const int lanelo = lane & 15, lanehi = (lane >> 4) & 1;
  const int mt = blockIdx.x;
  const int rowsLeft = M - mt * 16;
  const unsigned long long gbase =
      (unsigned long long)(uintptr_t)(flow + (size_t)mt * 16 * FDIM);
  const unsigned int lb0 = (unsigned int)(uintptr_t)&As[0][0];
  const unsigned int lb1 = (unsigned int)(uintptr_t)&As[1][0];
  const unsigned int rv = rowsLeft > 0 ? (unsigned int)rowsLeft : 0u;

  if (wave == 0) tdm_load_tile(gbase, lb0, rv);

  v8f acc0 = {}, acc1 = {};
  const int NCHK = FDIM / KCH;   // 16
  const int n0 = (wave * 2) * 16 + lanelo;
  for (int ch = 0; ch < NCHK; ++ch) {
    if (wave == 0) {
      if (ch + 1 < NCHK) {
        tdm_load_tile(gbase + (unsigned long long)(ch + 1) * KCH * 4,
                      ((ch + 1) & 1) ? lb1 : lb0, rv);
        __builtin_amdgcn_s_wait_tensorcnt(1);   // previous chunk complete
      } else {
        __builtin_amdgcn_s_wait_tensorcnt(0);   // last chunk complete
      }
    }
    __syncthreads();
    const float* Ab = &As[ch & 1][0];
    for (int kt = 0; kt < KCH / 32; ++kt) {
      const int kk = kt * 32 + lanehi * 16;
      const float* ar = Ab + lanelo * 258 + kk;   // padded row stride
      v16bf a;
#pragma unroll
      for (int i = 0; i < 8; ++i) {
        a[i]     = (__bf16)ar[i];
        a[8 + i] = (__bf16)ar[8 + i];
      }
      const int kglob = ch * KCH + kt * 32 + lanehi * 16;
      const v8bf* bp0 = (const v8bf*)(WrgbT + (size_t)n0 * FDIM + kglob);
      const v8bf* bp1 = (const v8bf*)(WrgbT + (size_t)(n0 + 16) * FDIM + kglob);
      const v16bf b0 = cat8(bp0[0], bp0[1]);
      const v16bf b1 = cat8(bp1[0], bp1[1]);
      acc0 = __builtin_amdgcn_wmma_f32_16x16x32_bf16(false, a, false, b0,
                                                     (short)0, acc0, false, false);
      acc1 = __builtin_amdgcn_wmma_f32_16x16x32_bf16(false, a, false, b1,
                                                     (short)0, acc1, false, false);
    }
    __syncthreads();
  }
  const float bias0 = brgb[n0], bias1 = brgb[n0 + 16];
#pragma unroll
  for (int r = 0; r < 8; ++r) {
    const int m = mt * 16 + r + 8 * lanehi;
    if (m < M) {
      const float v0 = acc0[r] + bias0;
      const float v1 = acc1[r] + bias1;
      RGB[(size_t)m * HID + n0]      = v0 > 0.f ? v0 : 0.f;
      RGB[(size_t)m * HID + n0 + 16] = v1 > 0.f ? v1 : 0.f;
    }
  }
}

// ---------------------------------------------------------------------------
// Kernel 2: GX = comb @ Wx_g + bx_g    (M=5730, K=512, N=768)
// comb rows are a gather [rgb[j+1] || rgb[0]] -> cooperative LDS stage (bf16,
// bank-padded); 8 waves x 6 N-tiles per block.
// ---------------------------------------------------------------------------
__global__ __launch_bounds__(256)
void gx_gemm(const float* __restrict__ RGB, const __bf16* __restrict__ WxgT,
             const float* __restrict__ bxg, float* __restrict__ GX) {
  __shared__ __bf16 As2[16 * 520];       // 512 + 8 pad per row
  const int M = T_STEPS * NBX;
  const int tid = threadIdx.x;
  const int wave = tid >> 5, lane = tid & 31;
  const int lanelo = lane & 15, lanehi = (lane >> 4) & 1;
  const int mt = blockIdx.x;
  for (int i = tid; i < 16 * 512; i += 256) {
    const int r = i >> 9, c = i & 511;
    int row = mt * 16 + r; row = row < M ? row : M - 1;
    const int t = row / NBX, j = row % NBX;
    const float v = (c < HID)
        ? RGB[(size_t)(t * NPOV + j + 1) * HID + c]
        : RGB[(size_t)(t * NPOV) * HID + (c - HID)];
    As2[r * 520 + c] = (__bf16)v;
  }
  __syncthreads();

  v8f acc[6] = {};
  for (int kt = 0; kt < (2 * HID) / 32; ++kt) {
    const int kk = kt * 32 + lanehi * 16;
    const v16bf a = cat8(*(const v8bf*)&As2[lanelo * 520 + kk],
                         *(const v8bf*)&As2[lanelo * 520 + kk + 8]);
    const int kb = kt * 32 + lanehi * 16;
#pragma unroll
    for (int u = 0; u < 6; ++u) {
      const int nc = (wave * 6 + u) * 16 + lanelo;
      const v8bf* bp = (const v8bf*)(WxgT + (size_t)nc * (2 * HID) + kb);
      const v16bf b = cat8(bp[0], bp[1]);
      acc[u] = __builtin_amdgcn_wmma_f32_16x16x32_bf16(false, a, false, b,
                                                       (short)0, acc[u], false, false);
    }
  }
#pragma unroll
  for (int u = 0; u < 6; ++u) {
    const int nc = (wave * 6 + u) * 16 + lanelo;
    const float bias = bxg[nc];
#pragma unroll
    for (int r = 0; r < 8; ++r) {
      const int m = mt * 16 + r + 8 * lanehi;
      if (m < M)
        GX[(size_t)m * (3 * HID) + nc] = acc[u][r] + bias;
    }
  }
}

// ---------------------------------------------------------------------------
// Kernel 2b: per-(t,j) small features: bfeat, dfeat -> xc -> gxc
// ---------------------------------------------------------------------------
__global__ __launch_bounds__(128)
void small_feats(const float* __restrict__ y,
                 const float* __restrict__ Wbb, const float* __restrict__ bbb,
                 const float* __restrict__ Wdd, const float* __restrict__ bdd,
                 const float* __restrict__ Wproj, const float* __restrict__ bproj,
                 const float* __restrict__ Wxc, const float* __restrict__ bxc,
                 float* __restrict__ BF, float* __restrict__ GXC) {
  const int bid = blockIdx.x;
  const int t = bid / NBX, j = bid % NBX;
  const float* yr = y + (size_t)(t * NBX + j) * 13;
  __shared__ float in5[5], in6[6], dfeat[HID], xc[COR];
  if (threadIdx.x == 0) {
    in5[0] = yr[1] / 1920.f; in5[1] = yr[2] / 1200.f;
    in5[2] = yr[3] / 1920.f; in5[3] = yr[4] / 1200.f;
    in5[4] = yr[6];
    for (int q = 0; q < 6; ++q) in6[q] = yr[7 + q];
  }
  __syncthreads();
  for (int c = threadIdx.x; c < HID; c += 128) {
    float b = bbb[c], d = bdd[c];
    for (int q = 0; q < 5; ++q) b += in5[q] * Wbb[q * HID + c];
    for (int q = 0; q < 6; ++q) d += in6[q] * Wdd[q * HID + c];
    BF[(size_t)(t * NBX + j) * HID + c] = b > 0.f ? b : 0.f;
    dfeat[c] = d > 0.f ? d : 0.f;
  }
  __syncthreads();
  for (int o = threadIdx.x; o < COR; o += 128) {
    float s = bproj[o];
    for (int c = 0; c < HID; ++c) s += dfeat[c] * Wproj[c * COR + o];
    xc[o] = s;
  }
  __syncthreads();
  for (int o = threadIdx.x; o < 3 * COR; o += 128) {
    float s = bxc[o];
    for (int q = 0; q < COR; ++q) s += xc[q] * Wxc[q * (3 * COR) + o];
    GXC[(size_t)(t * NBX + j) * (3 * COR) + o] = s;
  }
}

// ---------------------------------------------------------------------------
// Kernel 3: persistent single-workgroup recurrence (191 steps).
// h state bf16 in LDS (row stride 264 -> conflict-free A fragments);
// h @ Wh_g via WMMA with pre-transposed bf16 weights.
// ---------------------------------------------------------------------------
#define HSTR 264
__global__ __launch_bounds__(256)
void scan_kernel(const float* __restrict__ y,
                 const float* __restrict__ GX, const float* __restrict__ GXC,
                 const float* __restrict__ BF, const __bf16* __restrict__ WhgT,
                 const float* __restrict__ bhg,
                 const float* __restrict__ Whc, const float* __restrict__ bhc,
                 const float* __restrict__ Wout, const float* __restrict__ bout,
                 const float* __restrict__ Wpbox, const float* __restrict__ bpbox,
                 float* __restrict__ out) {
  __shared__ __bf16 sh_h[32 * HSTR];
  __shared__ __bf16 sh_hn[32 * HSTR];
  __shared__ float  sh_hc[NBX * COR];
  __shared__ float  sh_hcn[NBX * COR];
  __shared__ float  sh_s[NBX * NBX];
  __shared__ float  sh_act[NBX];
  __shared__ float  red[256];

  const int tid = threadIdx.x;
  const int wave = tid >> 5, lane = tid & 31;
  const int lanelo = lane & 15, lanehi = (lane >> 4) & 1;

  for (int i = tid; i < 32 * HSTR; i += 256) { sh_h[i] = (__bf16)0.f; sh_hn[i] = (__bf16)0.f; }
  for (int i = tid; i < NBX * COR; i += 256) { sh_hc[i] = 0.f; sh_hcn[i] = 0.f; }
  float bacc = 0.f, oacc = 0.f;
  __syncthreads();

  for (int t = 0; t < T_STEPS; ++t) {
    if (tid < NBX)
      sh_act[tid] = (y[(size_t)(t * NBX + tid) * 13] != 0.f) ? 1.f : 0.f;
    __syncthreads();

    // --- hc GRU (tiny, VALU) ---
    for (int idx = tid; idx < NBX * COR; idx += 256) {
      const int j = idx >> 5, c = idx & 31;
      float hr = bhc[c], hz = bhc[COR + c], hn = bhc[2 * COR + c];
      for (int q = 0; q < COR; ++q) {
        const float hq = sh_hc[j * COR + q];
        hr += hq * Whc[q * 3 * COR + c];
        hz += hq * Whc[q * 3 * COR + COR + c];
        hn += hq * Whc[q * 3 * COR + 2 * COR + c];
      }
      const size_t gb = (size_t)(t * NBX + j) * 3 * COR;
      const float rg = sigf(GXC[gb + c] + hr);
      const float zg = sigf(GXC[gb + COR + c] + hz);
      const float ng = tanhf(GXC[gb + 2 * COR + c] + rg * hn);
      sh_hcn[idx] = (1.f - zg) * ng + zg * sh_hc[idx];
    }

    // --- h GRU via WMMA (3 gate tiles per task) ---
    for (int task = wave; task < 32; task += 8) {
      const int mt = task >> 4;
      const int ct = task & 15;
      const int n = ct * 16 + lanelo;
      const int arow = mt * 16 + lanelo;
      v8f dr = {}, dz = {}, dn = {};
      for (int kt = 0; kt < HID / 32; ++kt) {
        const int k0 = kt * 32 + lanehi * 16;
        const v16bf a = cat8(*(const v8bf*)&sh_h[arow * HSTR + k0],
                             *(const v8bf*)&sh_h[arow * HSTR + k0 + 8]);
        const int kb = kt * 32 + lanehi * 16;
        const v8bf* pr = (const v8bf*)(WhgT + (size_t)n * HID + kb);
        const v8bf* pz = (const v8bf*)(WhgT + (size_t)(HID + n) * HID + kb);
        const v8bf* pn = (const v8bf*)(WhgT + (size_t)(2 * HID + n) * HID + kb);
        const v16bf br = cat8(pr[0], pr[1]);
        const v16bf bz = cat8(pz[0], pz[1]);
        const v16bf bn = cat8(pn[0], pn[1]);
        dr = __builtin_amdgcn_wmma_f32_16x16x32_bf16(false, a, false, br, (short)0, dr, false, false);
        dz = __builtin_amdgcn_wmma_f32_16x16x32_bf16(false, a, false, bz, (short)0, dz, false, false);
        dn = __builtin_amdgcn_wmma_f32_16x16x32_bf16(false, a, false, bn, (short)0, dn, false, false);
      }
#pragma unroll
      for (int r = 0; r < 8; ++r) {
        const int m = mt * 16 + r + 8 * lanehi;
        if (m < NBX) {
          const size_t gb = (size_t)(t * NBX + m) * (3 * HID);
          const float rg = sigf(GX[gb + n] + dr[r] + bhg[n]);
          const float zg = sigf(GX[gb + HID + n] + dz[r] + bhg[HID + n]);
          const float ng = tanhf(GX[gb + 2 * HID + n] + rg * (dn[r] + bhg[2 * HID + n]));
          const float hold = (float)sh_h[m * HSTR + n];
          sh_hn[m * HSTR + n] = (__bf16)((1.f - zg) * ng + zg * hold);
        }
      }
    }
    __syncthreads();

    // --- losses (pbox + logit) ---
    for (int idx = tid; idx < NBX * 37; idx += 256) {
      const int j = idx / 37, o = idx % 37;
      const float a = sh_act[j];
      const __bf16* hrow = sh_hn + j * HSTR;
      const float* bfr = BF + (size_t)(t * NBX + j) * HID;
      const float* hcr = sh_hcn + j * COR;
      float s = 0.f;
      if (o < 36) {
        for (int k = 0; k < HID; ++k) s += (float)hrow[k] * Wpbox[k * 36 + o];
        for (int k = 0; k < HID; ++k) s += bfr[k] * Wpbox[(HID + k) * 36 + o];
        for (int k = 0; k < COR; ++k) s += hcr[k] * Wpbox[(2 * HID + k) * 36 + o];
        s += bpbox[o];
        const int i9 = o >> 2, c4 = o & 3;
        const float norm4[4] = {1920.f, 1200.f, 1920.f, 1200.f};
        const float fut = y[(size_t)((t + 1 + i9) * NBX + j) * 13 + 1 + c4] / norm4[c4];
        const float d = s - fut, ad = fabsf(d);
        const float hub = ad < 1.f ? 0.5f * d * d : ad - 0.5f;
        bacc += hub * (1.f / 36.f) * a;
      } else {
        for (int k = 0; k < HID; ++k) s += (float)hrow[k] * Wout[k];
        for (int k = 0; k < HID; ++k) s += bfr[k] * Wout[HID + k];
        for (int k = 0; k < COR; ++k) s += hcr[k] * Wout[2 * HID + k];
        s += bout[0];
        const float tgt = y[(size_t)(t * NBX + j) * 13 + 5];
        const float bce = fmaxf(s, 0.f) - s * tgt + log1pf(__expf(-fabsf(s)));
        oacc += 0.05f * bce * a;
      }
    }
    __syncthreads();

    // --- h self-attention (h2 = act ? h_n : h) ---
    for (int idx = tid; idx < NBX * NBX; idx += 256) {
      const int i = idx / NBX, j = idx % NBX;
      const __bf16* hi = (sh_act[i] != 0.f ? sh_hn : sh_h) + i * HSTR;
      const __bf16* hj = (sh_act[j] != 0.f ? sh_hn : sh_h) + j * HSTR;
      float s = 0.f;
      for (int k = 0; k < HID; ++k) s += (float)hi[k] * (float)hj[k];
      s *= (1.f / 16.f);
      if (sh_act[j] == 0.f) s = -1e9f;
      sh_s[idx] = s;
    }
    __syncthreads();
    if (tid < NBX) {
      float mx = -1e30f;
      for (int j = 0; j < NBX; ++j) mx = fmaxf(mx, sh_s[tid * NBX + j]);
      float ssum = 0.f;
      for (int j = 0; j < NBX; ++j) {
        const float e = __expf(sh_s[tid * NBX + j] - mx);
        sh_s[tid * NBX + j] = e; ssum += e;
      }
      const float inv = 1.f / ssum;
      for (int j = 0; j < NBX; ++j) sh_s[tid * NBX + j] *= inv;
    }
    __syncthreads();
    float hout[4][8];
#pragma unroll
    for (int rr = 0; rr < 4; ++rr)
#pragma unroll
      for (int c = 0; c < 8; ++c) hout[rr][c] = 0.f;
#pragma unroll
    for (int rr = 0; rr < 4; ++rr) {
      const int i = wave + rr * 8;
      if (i < NBX && sh_act[i] != 0.f) {
        for (int j = 0; j < NBX; ++j) {
          const float aij = sh_s[i * NBX + j];
          const __bf16* hj = (sh_act[j] != 0.f ? sh_hn : sh_h) + j * HSTR + lane * 8;
#pragma unroll
          for (int c = 0; c < 8; ++c) hout[rr][c] += aij * (float)hj[c];
        }
      }
    }
    __syncthreads();
#pragma unroll
    for (int rr = 0; rr < 4; ++rr) {
      const int i = wave + rr * 8;
      if (i < NBX && sh_act[i] != 0.f)
#pragma unroll
        for (int c = 0; c < 8; ++c)
          sh_h[i * HSTR + lane * 8 + c] = (__bf16)hout[rr][c];
    }
    __syncthreads();

    // --- hc self-attention ---
    for (int idx = tid; idx < NBX * NBX; idx += 256) {
      const int i = idx / NBX, j = idx % NBX;
      const float* hi = (sh_act[i] != 0.f ? sh_hcn : sh_hc) + i * COR;
      const float* hj = (sh_act[j] != 0.f ? sh_hcn : sh_hc) + j * COR;
      float s = 0.f;
      for (int k = 0; k < COR; ++k) s += hi[k] * hj[k];
      s *= 0.17677669529f;
      if (sh_act[j] == 0.f) s = -1e9f;
      sh_s[idx] = s;
    }
    __syncthreads();
    if (tid < NBX) {
      float mx = -1e30f;
      for (int j = 0; j < NBX; ++j) mx = fmaxf(mx, sh_s[tid * NBX + j]);
      float ssum = 0.f;
      for (int j = 0; j < NBX; ++j) {
        const float e = __expf(sh_s[tid * NBX + j] - mx);
        sh_s[tid * NBX + j] = e; ssum += e;
      }
      const float inv = 1.f / ssum;
      for (int j = 0; j < NBX; ++j) sh_s[tid * NBX + j] *= inv;
    }
    __syncthreads();
    float cout[4] = {0.f, 0.f, 0.f, 0.f};
    for (int rr = 0; rr < 4; ++rr) {
      const int idx = tid + rr * 256;
      if (idx < NBX * COR) {
        const int i = idx >> 5, c = idx & 31;
        if (sh_act[i] != 0.f) {
          float s = 0.f;
          for (int j = 0; j < NBX; ++j) {
            const float* hj = (sh_act[j] != 0.f ? sh_hcn : sh_hc);
            s += sh_s[i * NBX + j] * hj[j * COR + c];
          }
          cout[rr] = s;
        }
      }
    }
    __syncthreads();
    for (int rr = 0; rr < 4; ++rr) {
      const int idx = tid + rr * 256;
      if (idx < NBX * COR) {
        const int i = idx >> 5;
        if (sh_act[i] != 0.f) sh_hc[idx] = cout[rr];
      }
    }
    __syncthreads();
  }

  red[tid] = bacc;
  __syncthreads();
  for (int s = 128; s > 0; s >>= 1) {
    if (tid < s) red[tid] += red[tid + s];
    __syncthreads();
  }
  const float btot = red[0];
  __syncthreads();
  red[tid] = oacc;
  __syncthreads();
  for (int s = 128; s > 0; s >>= 1) {
    if (tid < s) red[tid] += red[tid + s];
    __syncthreads();
  }
  if (tid == 0) out[0] = red[0] + 1.3f * btot;
}

// ---------------------------------------------------------------------------
extern "C" void kernel_launch(void* const* d_in, const int* in_sizes, int n_in,
                              void* d_out, int out_size, void* d_ws, size_t ws_size,
                              hipStream_t stream) {
  (void)in_sizes; (void)n_in; (void)out_size; (void)ws_size;
  const float* y     = (const float*)d_in[1];
  const float* flow  = (const float*)d_in[3];
  const float* Wrgb  = (const float*)d_in[4];
  const float* brgb  = (const float*)d_in[5];
  const float* Wbb   = (const float*)d_in[6];
  const float* bbb   = (const float*)d_in[7];
  const float* Wdd   = (const float*)d_in[8];
  const float* bdd   = (const float*)d_in[9];
  const float* Wproj = (const float*)d_in[10];
  const float* bproj = (const float*)d_in[11];
  const float* Wxc   = (const float*)d_in[12];
  const float* Whc   = (const float*)d_in[13];
  const float* bxc   = (const float*)d_in[14];
  const float* bhc   = (const float*)d_in[15];
  const float* Wxg   = (const float*)d_in[16];
  const float* Whg   = (const float*)d_in[17];
  const float* bxg   = (const float*)d_in[18];
  const float* bhg   = (const float*)d_in[19];
  const float* Wout  = (const float*)d_in[20];
  const float* bout  = (const float*)d_in[21];
  const float* Wpbox = (const float*)d_in[22];
  const float* bpbox = (const float*)d_in[23];

  float* ws  = (float*)d_ws;
  float* RGB = ws + OFF_RGB;
  float* GX  = ws + OFF_GX;
  float* BF  = ws + OFF_BF;
  float* GXC = ws + OFF_GXC;
  __bf16* WT = (__bf16*)(ws + OFF_WT);

  prep_weights<<<dim3(6400), 256, 0, stream>>>(Wrgb, Wxg, Whg, WT);
  rgb_gemm<<<dim3(371), 256, 0, stream>>>(flow, WT + WB_RGBT, brgb, RGB);
  gx_gemm<<<dim3(359), 256, 0, stream>>>(RGB, WT + WB_XGT, bxg, GX);
  small_feats<<<dim3(T_STEPS * NBX), 128, 0, stream>>>(
      y, Wbb, bbb, Wdd, bdd, Wproj, bproj, Wxc, bxc, BF, GXC);
  scan_kernel<<<dim3(1), 256, 0, stream>>>(
      y, GX, GXC, BF, WT + WB_HGT, bhg, Whc, bhc, Wout, bout, Wpbox, bpbox,
      (float*)d_out);
}
